// Transformer_10230612099073
// MI455X (gfx1250) — compile-verified, matrix operand order checked by reading, not run on hardware
//
#include <hip/hip_runtime.h>
#include <hip/hip_bf16.h>

// ---------------------------------------------------------------------------
// Model constants (match reference)
// ---------------------------------------------------------------------------
#define D_   512
#define H_   8
#define HD_  64
#define F_   2048
#define S_   1024
#define T_   1024
#define B_   4
#define NROW 4096          // S_*B_ == T_*B_
#define LD_  (B_ * D_)     // row stride (elements) between consecutive t
#define SMAX 1024          // keys per attention row (16x1024 f32 = 64KB LDS)
#define EPS_ 1e-5f

typedef __attribute__((ext_vector_type(16))) __bf16 v16bf;
typedef __attribute__((ext_vector_type(8)))  float  v8f;

// f32 -> bf16 (RNE), scalar
__device__ __forceinline__ unsigned short f2bf_bits(float f) {
    unsigned u = __builtin_bit_cast(unsigned, f);
    unsigned r = (u + 0x7FFFu + ((u >> 16) & 1u)) >> 16;
    return (unsigned short)r;
}
__device__ __forceinline__ __bf16 f2bf(float f) {
    unsigned short s = f2bf_bits(f);
    return __builtin_bit_cast(__bf16, s);
}

// Load a 16x32 bf16 A/B fragment lane-slice from a K-contiguous bf16 row.
// rowp: lane's row start (already offset by k0). kh = (lane>>4)*8.
// Elements 0..7 = K[kh..kh+7], 8..15 = K[kh+16..kh+23]  (two b128 loads).
__device__ __forceinline__ v16bf ldfrag(const __bf16* __restrict__ rowp, int kh) {
    union { uint4 q[2]; v16bf v; } u;
    u.q[0] = *(const uint4*)(rowp + kh);
    u.q[1] = *(const uint4*)(rowp + kh + 16);
    return u.v;
}

__device__ __forceinline__ v8f wmma_bf16(v16bf a, v16bf b, v8f c) {
    return __builtin_amdgcn_wmma_f32_16x16x32_bf16(
        false, a, false, b, (short)0, c, false, false);
}

// ---------------------------------------------------------------------------
// GEMM: Y[M,N] = A[M,K](bf16) @ W[N,K](bf16)^T + bias[N]  (opt ReLU)
// Per-wave tile 32(M) x 64(N): 2x4 accumulators, 8 WMMA per 32-K step.
// 8 waves/block -> 128x128 block tile. Grid: (N/128, M/128).
// out_bf16: store bf16 (consumer is another WMMA kernel) else f32.
// ---------------------------------------------------------------------------
__global__ void __launch_bounds__(256)
wmma_gemm(const __bf16* __restrict__ A, const __bf16* __restrict__ W,
          const float* __restrict__ bias, void* __restrict__ Yout,
          int M, int N, int K, int relu, int out_bf16)
{
    const int lane = threadIdx.x & 31;
    const int wave = threadIdx.x >> 5;
    const int m0 = blockIdx.y * 128 + (wave & 3) * 32;
    const int n0 = blockIdx.x * 128 + (wave >> 2) * 64;
    const int kh = (lane >> 4) << 3;

    const __bf16* __restrict__ a0 = A + (size_t)(m0 + (lane & 15)) * K;
    const __bf16* __restrict__ a1 = a0 + (size_t)16 * K;
    const __bf16* __restrict__ b0 = W + (size_t)(n0 + (lane & 15)) * K;
    const __bf16* __restrict__ b1 = b0 + (size_t)16 * K;
    const __bf16* __restrict__ b2 = b0 + (size_t)32 * K;
    const __bf16* __restrict__ b3 = b0 + (size_t)48 * K;

    v8f acc[2][4] = {};
    for (int k0 = 0; k0 < K; k0 += 32) {
        if (k0 + 64 < K) {          // warm L2/L0 for the next-next step
            __builtin_prefetch(a0 + k0 + 64, 0, 1);
            __builtin_prefetch(b0 + k0 + 64, 0, 1);
        }
        v16bf Af0 = ldfrag(a0 + k0, kh);
        v16bf Af1 = ldfrag(a1 + k0, kh);
        v16bf Bf0 = ldfrag(b0 + k0, kh);
        v16bf Bf1 = ldfrag(b1 + k0, kh);
        v16bf Bf2 = ldfrag(b2 + k0, kh);
        v16bf Bf3 = ldfrag(b3 + k0, kh);
        acc[0][0] = wmma_bf16(Af0, Bf0, acc[0][0]);
        acc[0][1] = wmma_bf16(Af0, Bf1, acc[0][1]);
        acc[0][2] = wmma_bf16(Af0, Bf2, acc[0][2]);
        acc[0][3] = wmma_bf16(Af0, Bf3, acc[0][3]);
        acc[1][0] = wmma_bf16(Af1, Bf0, acc[1][0]);
        acc[1][1] = wmma_bf16(Af1, Bf1, acc[1][1]);
        acc[1][2] = wmma_bf16(Af1, Bf2, acc[1][2]);
        acc[1][3] = wmma_bf16(Af1, Bf3, acc[1][3]);
    }

    const int rh = (lane >> 4) << 3;      // +0 or +8 row offset
    const int cl = lane & 15;
#pragma unroll
    for (int i = 0; i < 2; ++i) {
#pragma unroll
        for (int j = 0; j < 4; ++j) {
            const int col = n0 + j * 16 + cl;
            const float bb = bias ? bias[col] : 0.0f;
#pragma unroll
            for (int jj = 0; jj < 8; ++jj) {
                const int row = m0 + i * 16 + rh + jj;
                float v = acc[i][j][jj] + bb;
                if (relu) v = fmaxf(v, 0.0f);
                if (out_bf16) ((__bf16*)Yout)[(size_t)row * N + col] = f2bf(v);
                else          ((float*)Yout)[(size_t)row * N + col] = v;
            }
        }
    }
}

// ---------------------------------------------------------------------------
// Attention for one (16-query tile, b, h). One wave per block.
// Q/K: bf16 [L, B_*D_] (head slice at b*D_+h*HD_).
// VT:  bf16 [B_*H_][HD_][Lk]  (pre-transposed -> contiguous B fragments).
// O:   bf16 [L, B_*D_].
// Scores (16 x SMAX f32) in LDS with XOR bank swizzle.
// ---------------------------------------------------------------------------
__global__ void __launch_bounds__(32)
wmma_attention(const __bf16* __restrict__ Qb, const __bf16* __restrict__ Kb,
               const __bf16* __restrict__ VT, __bf16* __restrict__ Ob,
               int Lq, int Lk, int causal)
{
    __shared__ float sc[16 * SMAX];

    const int lane = threadIdx.x;
    const int t0   = blockIdx.x * 16;
    const int bh   = blockIdx.y;
    const int b    = bh / H_;
    const int h    = bh % H_;
    const size_t hoff = (size_t)b * D_ + (size_t)h * HD_;

    const __bf16* __restrict__ Q = Qb + hoff;
    const __bf16* __restrict__ K = Kb + hoff;
    const __bf16* __restrict__ Vt = VT + (size_t)bh * HD_ * SMAX;
    __bf16*       __restrict__ O = Ob + hoff;

    const int kh = (lane >> 4) << 3;
    const float scale = 0.125f;                 // HD^-0.5

    const int rq = t0 + (lane & 15);
    const __bf16* qrow = Q + (size_t)rq * LD_;
    v16bf qa0 = ldfrag(qrow +  0, kh);
    v16bf qa1 = ldfrag(qrow + 32, kh);

    // ---- scores = (Q K^T) * scale + mask ----
    for (int s0 = 0; s0 < Lk; s0 += 16) {
        const __bf16* krow = K + (size_t)(s0 + (lane & 15)) * LD_;
        v16bf kb0 = ldfrag(krow +  0, kh);
        v16bf kb1 = ldfrag(krow + 32, kh);
        v8f d = {};
        d = wmma_bf16(qa0, kb0, d);
        d = wmma_bf16(qa1, kb1, d);
        const int s = s0 + (lane & 15);
#pragma unroll
        for (int j = 0; j < 8; ++j) {
            const int m = j + ((lane >> 4) << 3);
            float v = d[j] * scale;
            if (causal && s > t0 + m) v = -1e9f;
            sc[m * SMAX + (s ^ (m << 2))] = v;
        }
    }
    __syncthreads();

    // ---- row softmax (swizzle is a within-row permutation) ----
    for (int m = 0; m < 16; ++m) {
        float* row = sc + m * SMAX;
        float mx = -3.4e38f;
        for (int s = lane; s < Lk; s += 32) mx = fmaxf(mx, row[s]);
#pragma unroll
        for (int off = 16; off; off >>= 1) mx = fmaxf(mx, __shfl_xor(mx, off, 32));
        float sum = 0.0f;
        for (int s = lane; s < Lk; s += 32) { float e = __expf(row[s] - mx); row[s] = e; sum += e; }
#pragma unroll
        for (int off = 16; off; off >>= 1) sum += __shfl_xor(sum, off, 32);
        const float inv = 1.0f / sum;
        for (int s = lane; s < Lk; s += 32) row[s] *= inv;
    }
    __syncthreads();

    // ---- O = P @ V : contract over s, 4 column chunks of 16 ----
    v8f o0 = {}, o1 = {}, o2 = {}, o3 = {};
    const int m = lane & 15;
    for (int s0 = 0; s0 < Lk; s0 += 32) {
        v16bf p;
#pragma unroll
        for (int i = 0; i < 8; ++i) {
            p[i]     = f2bf(sc[m * SMAX + ((s0 + kh + i)      ^ (m << 2))]);
            p[i + 8] = f2bf(sc[m * SMAX + ((s0 + kh + 16 + i) ^ (m << 2))]);
        }
        // B column n = VT row (d = nc*16 + n), contiguous in s
        const __bf16* v0 = Vt + (size_t)( 0 + m) * SMAX + s0;
        const __bf16* v1 = Vt + (size_t)(16 + m) * SMAX + s0;
        const __bf16* v2 = Vt + (size_t)(32 + m) * SMAX + s0;
        const __bf16* v3 = Vt + (size_t)(48 + m) * SMAX + s0;
        o0 = wmma_bf16(p, ldfrag(v0, kh), o0);
        o1 = wmma_bf16(p, ldfrag(v1, kh), o1);
        o2 = wmma_bf16(p, ldfrag(v2, kh), o2);
        o3 = wmma_bf16(p, ldfrag(v3, kh), o3);
    }

    const int rbase = t0 + ((lane >> 4) << 3);
    const int cl    = lane & 15;
#pragma unroll
    for (int j = 0; j < 8; ++j) {
        __bf16* orow = O + (size_t)(rbase + j) * LD_;
        orow[ 0 + cl] = f2bf(o0[j]);
        orow[16 + cl] = f2bf(o1[j]);
        orow[32 + cl] = f2bf(o2[j]);
        orow[48 + cl] = f2bf(o3[j]);
    }
}

// ---------------------------------------------------------------------------
// Elementwise f32 -> bf16 (8 elems/thread, packed dword stores)
// ---------------------------------------------------------------------------
__global__ void __launch_bounds__(256)
cvt_bf16(const float* __restrict__ X, __bf16* __restrict__ Y, long n)
{
    long i = ((long)blockIdx.x * 256 + threadIdx.x) * 8;
    if (i + 8 > n) return;
    float4 a = *(const float4*)(X + i);
    float4 b = *(const float4*)(X + i + 4);
    uint4 o;
    o.x = (unsigned)f2bf_bits(a.x) | ((unsigned)f2bf_bits(a.y) << 16);
    o.y = (unsigned)f2bf_bits(a.z) | ((unsigned)f2bf_bits(a.w) << 16);
    o.z = (unsigned)f2bf_bits(b.x) | ((unsigned)f2bf_bits(b.y) << 16);
    o.w = (unsigned)f2bf_bits(b.z) | ((unsigned)f2bf_bits(b.w) << 16);
    *(uint4*)(Y + i) = o;
}

// ---------------------------------------------------------------------------
// V (f32 [L, B_*D_]) -> VT (bf16 [B_*H_][HD_][L])
// Grid: (L/256, HD_, B_*H_)
// ---------------------------------------------------------------------------
__global__ void __launch_bounds__(256)
transpose_v(const float* __restrict__ Vb, __bf16* __restrict__ VT, int L)
{
    const int s  = blockIdx.x * 256 + threadIdx.x;
    const int d  = blockIdx.y;
    const int bh = blockIdx.z;
    const int b  = bh / H_;
    const int h  = bh % H_;
    float v = Vb[(size_t)s * LD_ + (size_t)b * D_ + h * HD_ + d];
    VT[((size_t)bh * HD_ + d) * L + s] = f2bf(v);
}

// ---------------------------------------------------------------------------
// y = LayerNorm(x (+ r)) * g + b ; one row (D_=512) per 256-thread block.
// ---------------------------------------------------------------------------
__global__ void __launch_bounds__(256)
add_layernorm(const float* __restrict__ X, const float* __restrict__ R,
              const float* __restrict__ g, const float* __restrict__ bta,
              float* __restrict__ Y)
{
    __shared__ float ssum[8], ssq[8];
    const int row  = blockIdx.x;
    const int tid  = threadIdx.x;
    const int lane = tid & 31;
    const int wv   = tid >> 5;

    const float* x = X + (size_t)row * D_;
    const float* r = R ? R + (size_t)row * D_ : nullptr;

    float t0v = x[tid]       + (r ? r[tid]       : 0.0f);
    float t1v = x[tid + 256] + (r ? r[tid + 256] : 0.0f);

    float s = t0v + t1v;
    float q = t0v * t0v + t1v * t1v;
#pragma unroll
    for (int off = 16; off; off >>= 1) {
        s += __shfl_xor(s, off, 32);
        q += __shfl_xor(q, off, 32);
    }
    if (lane == 0) { ssum[wv] = s; ssq[wv] = q; }
    __syncthreads();
    float S = 0.0f, Q = 0.0f;
#pragma unroll
    for (int w = 0; w < 8; ++w) { S += ssum[w]; Q += ssq[w]; }

    const float mean = S * (1.0f / D_);
    const float var  = Q * (1.0f / D_) - mean * mean;
    const float inv  = rsqrtf(var + EPS_);

    Y[(size_t)row * D_ + tid]       = (t0v - mean) * inv * g[tid]       + bta[tid];
    Y[(size_t)row * D_ + tid + 256] = (t1v - mean) * inv * g[tid + 256] + bta[tid + 256];
}

// ---------------------------------------------------------------------------
// Host orchestration
// ---------------------------------------------------------------------------
extern "C" void kernel_launch(void* const* d_in, const int* in_sizes, int n_in,
                              void* d_out, int out_size, void* d_ws, size_t ws_size,
                              hipStream_t stream)
{
    (void)in_sizes; (void)n_in; (void)out_size; (void)ws_size;
    int i = 0;
    const float* src = (const float*)d_in[i++];
    const float* tgt = (const float*)d_in[i++];
    const float* e_wq  = (const float*)d_in[i++]; const float* e_bq  = (const float*)d_in[i++];
    const float* e_wk  = (const float*)d_in[i++]; const float* e_bk  = (const float*)d_in[i++];
    const float* e_wv  = (const float*)d_in[i++]; const float* e_bv  = (const float*)d_in[i++];
    const float* e_wo  = (const float*)d_in[i++]; const float* e_bo  = (const float*)d_in[i++];
    const float* e_w1  = (const float*)d_in[i++]; const float* e_b1  = (const float*)d_in[i++];
    const float* e_w2  = (const float*)d_in[i++]; const float* e_b2  = (const float*)d_in[i++];
    const float* e_l1g = (const float*)d_in[i++]; const float* e_l1b = (const float*)d_in[i++];
    const float* e_l2g = (const float*)d_in[i++]; const float* e_l2b = (const float*)d_in[i++];
    const float* dsa_wq = (const float*)d_in[i++]; const float* dsa_bq = (const float*)d_in[i++];
    const float* dsa_wk = (const float*)d_in[i++]; const float* dsa_bk = (const float*)d_in[i++];
    const float* dsa_wv = (const float*)d_in[i++]; const float* dsa_bv = (const float*)d_in[i++];
    const float* dsa_wo = (const float*)d_in[i++]; const float* dsa_bo = (const float*)d_in[i++];
    const float* dca_wq = (const float*)d_in[i++]; const float* dca_bq = (const float*)d_in[i++];
    const float* dca_wk = (const float*)d_in[i++]; const float* dca_bk = (const float*)d_in[i++];
    const float* dca_wv = (const float*)d_in[i++]; const float* dca_bv = (const float*)d_in[i++];
    const float* dca_wo = (const float*)d_in[i++]; const float* dca_bo = (const float*)d_in[i++];
    const float* d_w1  = (const float*)d_in[i++]; const float* d_b1  = (const float*)d_in[i++];
    const float* d_w2  = (const float*)d_in[i++]; const float* d_b2  = (const float*)d_in[i++];
    const float* d_l1g = (const float*)d_in[i++]; const float* d_l1b = (const float*)d_in[i++];
    const float* d_l2g = (const float*)d_in[i++]; const float* d_l2b = (const float*)d_in[i++];
    const float* d_l3g = (const float*)d_in[i++]; const float* d_l3b = (const float*)d_in[i++];
    const float* enc_ng = (const float*)d_in[i++]; const float* enc_nb = (const float*)d_in[i++];
    const float* dec_ng = (const float*)d_in[i++]; const float* dec_nb = (const float*)d_in[i++];

    // ---------------- workspace carve-up ----------------
    const size_t SZ  = (size_t)NROW * D_;     // 2M elements
    const size_t WSQ = (size_t)D_ * D_;       // 256K
    const size_t WF  = (size_t)F_ * D_;       // 1M

    char* base = (char*)d_ws;
    float* X   = (float*)base;                 base += SZ * 4;   // activations (f32)
    float* T1  = (float*)base;                 base += SZ * 4;   // residual branch (f32)
    float* Vb  = (float*)base;                 base += SZ * 4;   // V pre-transpose (f32)
    float* MEM = (float*)base;                 base += SZ * 4;   // encoder memory (f32)
    __bf16* XBF  = (__bf16*)base;              base += SZ * 2;
    __bf16* QBF  = (__bf16*)base;              base += SZ * 2;
    __bf16* KBF  = (__bf16*)base;              base += SZ * 2;
    __bf16* ATBF = (__bf16*)base;              base += SZ * 2;
    __bf16* MEMBF= (__bf16*)base;              base += SZ * 2;
    __bf16* VTB  = (__bf16*)base;              base += SZ * 2;   // [B*H][64][1024]
    __bf16* HIDBF= (__bf16*)base;              base += (size_t)NROW * F_ * 2;
    __bf16* WB   = (__bf16*)base;              // weight pool (bf16)

    // weight pool offsets
    __bf16* w_e_wq = WB;                 __bf16* w_e_wk = w_e_wq + WSQ;
    __bf16* w_e_wv = w_e_wk + WSQ;       __bf16* w_e_wo = w_e_wv + WSQ;
    __bf16* w_e_w1 = w_e_wo + WSQ;       __bf16* w_e_w2 = w_e_w1 + WF;
    __bf16* w_sa_wq = w_e_w2 + WF;       __bf16* w_sa_wk = w_sa_wq + WSQ;
    __bf16* w_sa_wv = w_sa_wk + WSQ;     __bf16* w_sa_wo = w_sa_wv + WSQ;
    __bf16* w_ca_wq = w_sa_wo + WSQ;     __bf16* w_ca_wk = w_ca_wq + WSQ;
    __bf16* w_ca_wv = w_ca_wk + WSQ;     __bf16* w_ca_wo = w_ca_wv + WSQ;
    __bf16* w_d_w1 = w_ca_wo + WSQ;      __bf16* w_d_w2 = w_d_w1 + WF;

    auto cvt = [&](const float* src_, __bf16* dst_, size_t n) {
        cvt_bf16<<<(unsigned)(n / 2048), 256, 0, stream>>>(src_, dst_, (long)n);
    };
    auto gemm = [&](const __bf16* A, const __bf16* W, const float* b, void* Y,
                    int M, int N, int K, int relu, int obf) {
        dim3 g(N / 128, M / 128);
        wmma_gemm<<<g, 256, 0, stream>>>(A, W, b, Y, M, N, K, relu, obf);
    };
    auto attn = [&](const __bf16* q, const __bf16* k, const __bf16* vt, __bf16* o,
                    int Lq, int Lk, int causal) {
        dim3 g(Lq / 16, B_ * H_);
        wmma_attention<<<g, 32, 0, stream>>>(q, k, vt, o, Lq, Lk, causal);
    };
    auto mk_vt = [&](const float* v, __bf16* vt, int L) {
        dim3 g(L / 256, HD_, B_ * H_);
        transpose_v<<<g, 256, 0, stream>>>(v, vt, L);
    };
    auto lnorm = [&](const float* x, const float* r, const float* g_, const float* b_,
                     float* y) {
        add_layernorm<<<NROW, 256, 0, stream>>>(x, r, g_, b_, y);
    };

    // ---------------- weights -> bf16 (once per call) ----------------
    cvt(e_wq, w_e_wq, WSQ);  cvt(e_wk, w_e_wk, WSQ);
    cvt(e_wv, w_e_wv, WSQ);  cvt(e_wo, w_e_wo, WSQ);
    cvt(e_w1, w_e_w1, WF);   cvt(e_w2, w_e_w2, WF);
    cvt(dsa_wq, w_sa_wq, WSQ); cvt(dsa_wk, w_sa_wk, WSQ);
    cvt(dsa_wv, w_sa_wv, WSQ); cvt(dsa_wo, w_sa_wo, WSQ);
    cvt(dca_wq, w_ca_wq, WSQ); cvt(dca_wk, w_ca_wk, WSQ);
    cvt(dca_wv, w_ca_wv, WSQ); cvt(dca_wo, w_ca_wo, WSQ);
    cvt(d_w1, w_d_w1, WF);     cvt(d_w2, w_d_w2, WF);

    // ---------------- Encoder ----------------
    hipMemcpyAsync(X, src, SZ * sizeof(float), hipMemcpyDeviceToDevice, stream);
    for (int l = 0; l < 2; ++l) {
        cvt(X, XBF, SZ);
        gemm(XBF, w_e_wq, e_bq, QBF, NROW, D_, D_, 0, 1);
        gemm(XBF, w_e_wk, e_bk, KBF, NROW, D_, D_, 0, 1);
        gemm(XBF, w_e_wv, e_bv, Vb,  NROW, D_, D_, 0, 0);
        mk_vt(Vb, VTB, S_);
        attn(QBF, KBF, VTB, ATBF, S_, S_, 0);
        gemm(ATBF, w_e_wo, e_bo, T1, NROW, D_, D_, 0, 0);
        lnorm(X, T1, e_l1g, e_l1b, X);
        cvt(X, XBF, SZ);
        gemm(XBF, w_e_w1, e_b1, HIDBF, NROW, F_, D_, 1, 1);
        gemm(HIDBF, w_e_w2, e_b2, T1, NROW, D_, F_, 0, 0);
        lnorm(X, T1, e_l2g, e_l2b, X);
    }
    lnorm(X, nullptr, enc_ng, enc_nb, MEM);
    cvt(MEM, MEMBF, SZ);

    // ---------------- Decoder ----------------
    hipMemcpyAsync(X, tgt, SZ * sizeof(float), hipMemcpyDeviceToDevice, stream);
    for (int l = 0; l < 2; ++l) {
        // causal self-attention
        cvt(X, XBF, SZ);
        gemm(XBF, w_sa_wq, dsa_bq, QBF, NROW, D_, D_, 0, 1);
        gemm(XBF, w_sa_wk, dsa_bk, KBF, NROW, D_, D_, 0, 1);
        gemm(XBF, w_sa_wv, dsa_bv, Vb,  NROW, D_, D_, 0, 0);
        mk_vt(Vb, VTB, T_);
        attn(QBF, KBF, VTB, ATBF, T_, T_, 1);
        gemm(ATBF, w_sa_wo, dsa_bo, T1, NROW, D_, D_, 0, 0);
        lnorm(X, T1, d_l1g, d_l1b, X);
        // cross-attention vs encoder memory
        cvt(X, XBF, SZ);
        gemm(XBF,   w_ca_wq, dca_bq, QBF, NROW, D_, D_, 0, 1);
        gemm(MEMBF, w_ca_wk, dca_bk, KBF, NROW, D_, D_, 0, 1);
        gemm(MEMBF, w_ca_wv, dca_bv, Vb,  NROW, D_, D_, 0, 0);
        mk_vt(Vb, VTB, S_);
        attn(QBF, KBF, VTB, ATBF, T_, S_, 0);
        gemm(ATBF, w_ca_wo, dca_bo, T1, NROW, D_, D_, 0, 0);
        lnorm(X, T1, d_l2g, d_l2b, X);
        // FFN
        cvt(X, XBF, SZ);
        gemm(XBF, w_d_w1, d_b1, HIDBF, NROW, F_, D_, 1, 1);
        gemm(HIDBF, w_d_w2, d_b2, T1, NROW, D_, F_, 0, 0);
        lnorm(X, T1, d_l3g, d_l3b, X);
    }
    lnorm(X, nullptr, dec_ng, dec_nb, (float*)d_out);
}
